// EGNNLayer_58875411693668
// MI455X (gfx1250) — matrix operand order repaired and checked by reading, not compiled, not run on hardware
//
#include <hip/hip_runtime.h>
#include <hip/hip_bf16.h>

// ---------------------------------------------------------------------------
// EGNN layer for MI455X (gfx1250), wave32, bf16 WMMA (v_wmma_f32_16x16x32_bf16)
//
// Inputs (setup_inputs order):
//  0 h [N,128] f32       1 x [N,3] f32        2 edge_index [2,E] i64
//  3 ew1 [257,128]       4 eb1 [128]          5 ew2 [128,128]     6 eb2 [128]
//  7 nw1 [256,128]       8 nb1 [128]          9 nw2 [128,128]    10 nb2 [128]
// 11 cw1 [128,128]      12 cb1 [128]         13 cw2 [128,1]
// Output: h_new [N*128] ++ x_new [N*3]  (f32)
// ---------------------------------------------------------------------------

typedef __attribute__((ext_vector_type(16))) __bf16          v16bf;
typedef __attribute__((ext_vector_type(8)))  float           v8f;
typedef __attribute__((ext_vector_type(8)))  unsigned short  v8s;
typedef __attribute__((ext_vector_type(16))) unsigned short  v16s;
typedef __attribute__((ext_vector_type(4)))  unsigned short  v4s;

union V16U { v16s s; v16bf b; };

// Native bf16 conversion (RNE) -> v_cvt_*bf16* on gfx1250
static __device__ __forceinline__ unsigned short f2bf(float f) {
    return __builtin_bit_cast(unsigned short, (__bf16)f);
}

static __device__ __forceinline__ v4s pack4(float4 v) {
    v4s r;
    r[0] = f2bf(v.x); r[1] = f2bf(v.y); r[2] = f2bf(v.z); r[3] = f2bf(v.w);
    return r;
}

// Fast SiLU: v * rcp(1 + exp2(v * -log2(e))).  v_exp_f32 / v_rcp_f32 are
// TRANS-class ops that co-issue with VALU/WMMA; avoids the ~8-op precise
// f32 division expansion.
static __device__ __forceinline__ float silu(float v) {
    const float e = __builtin_amdgcn_exp2f(v * -1.442695040888963f);
    return v * __builtin_amdgcn_rcpf(1.0f + e);
}

// Assemble v16bf A/B fragment from two 8x16b chunks
static __device__ __forceinline__ v16bf cat16(v8s lo, v8s hi) {
    V16U u;
#pragma unroll
    for (int i = 0; i < 8; ++i) { u.s[i] = lo[i]; u.s[i + 8] = hi[i]; }
    return u.b;
}

// A fragment (16x32 bf16) from LDS, row-major with `stride` (in shorts).
// lane<16:  row=lane,    K = kt*32 + {0..7, 16..23}
// lane>=16: row=lane-16, K = kt*32 + {8..15, 24..31}
static __device__ __forceinline__ v16bf ldA(const unsigned short* base, int stride,
                                            int kt, int lane) {
    const int m = lane & 15, half = lane >> 4;
    const v8s lo = *(const v8s*)&base[m * stride + kt * 32 + half * 8];
    const v8s hi = *(const v8s*)&base[m * stride + kt * 32 + 16 + half * 8];
    return cat16(lo, hi);
}

// B fragment (32x16 bf16) from pre-swizzled weights: tile = kt*8 + nt,
// each lane owns 16 contiguous shorts.
static __device__ __forceinline__ v16bf ldB(const unsigned short* wt, int tile, int lane) {
    const unsigned short* p = &wt[(size_t)tile * 512 + lane * 16];
    const v8s lo = *(const v8s*)p;
    const v8s hi = *(const v8s*)(p + 8);
    return cat16(lo, hi);
}

static __device__ __forceinline__ v8f wmma_bf16(v16bf a, v16bf b, v8f c) {
    return __builtin_amdgcn_wmma_f32_16x16x32_bf16(false, a, false, b, (short)0, c,
                                                   false, false);
}

// ---------------------------------------------------------------------------
// Weight swizzle: fp32 [K,N] row-major  ->  bf16 B-operand tiles
// tile t = kt*(N/16)+nt, element (lane,i): k = kt*32 + (i>>3)*16 + (lane>>4)*8 + (i&7),
// n = nt*16 + (lane&15)
// ---------------------------------------------------------------------------
__global__ void egnn_swz_kernel(const float* __restrict__ W,
                                unsigned short* __restrict__ dst, int K, int Nn) {
    const int idx = blockIdx.x * blockDim.x + threadIdx.x;
    if (idx >= K * Nn) return;
    const int i    = idx & 15;
    const int lane = (idx >> 4) & 31;
    const int t    = idx >> 9;
    const int ntn  = Nn >> 4;
    const int kt = t / ntn, nt = t % ntn;
    const int k = kt * 32 + (i >> 3) * 16 + (lane >> 4) * 8 + (i & 7);
    const int n = nt * 16 + (lane & 15);
    dst[idx] = f2bf(W[(size_t)k * Nn + n]);
}

// ---------------------------------------------------------------------------
// Edge kernel: 16 edges per wave, 4 waves per block. E must be a multiple of 64.
// ---------------------------------------------------------------------------
#define EW 4          // waves per block
#define FS 264        // feat LDS row stride (shorts), 256 + pad
#define HS 136        // hidden LDS row stride (shorts), 128 + pad

__global__ __launch_bounds__(EW * 32) void egnn_edge_kernel(
    const float* __restrict__ h, const float* __restrict__ x,
    const long long* __restrict__ eidx, int E,
    const unsigned short* __restrict__ ew1s, const float* __restrict__ eb1,
    const unsigned short* __restrict__ ew2s, const float* __restrict__ eb2,
    const float* __restrict__ ew1r,              // ew1 row 256 (radial), f32 [128]
    const unsigned short* __restrict__ cw1s, const float* __restrict__ cb1,
    const float* __restrict__ cw2,               // f32 [128]
    float* __restrict__ mi_accum, float* __restrict__ x_accum) {

    __shared__ unsigned short featS[EW][16 * FS];
    __shared__ unsigned short hidS[EW][16 * HS];
    __shared__ float diffS[EW][16][4];
    __shared__ float radS[EW][16];
    __shared__ int   rowS[EW][16];
    __shared__ float cwS[EW][16];

    const int lane = threadIdx.x & 31;
    const int wid  = threadIdx.x >> 5;
    const int mrow = lane & 15, half = lane >> 4;
    const long long e0 = ((long long)blockIdx.x * EW + wid) * 16;

    unsigned short* feat = featS[wid];
    unsigned short* hid  = hidS[wid];

    // ---- per-edge scalars (lanes 0..15) --------------------------------
    int myrow = 0, mycol = 0;
    if (lane < 16) {
        const long long e = e0 + lane;
        myrow = (int)eidx[e];
        mycol = (int)eidx[(long long)E + e];
        float r2 = 0.0f;
#pragma unroll
        for (int d = 0; d < 3; ++d) {
            const float dx = x[(size_t)myrow * 3 + d] - x[(size_t)mycol * 3 + d];
            diffS[wid][lane][d] = dx;
            r2 += dx * dx;
        }
        radS[wid][lane] = r2;
        rowS[wid][lane] = myrow;
    }
    __builtin_amdgcn_wave_barrier();

    // ---- gather h[row] || h[col] -> bf16 feat tile [16 x 256] ----------
#pragma unroll 4
    for (int m = 0; m < 16; ++m) {
        const int r = __shfl(myrow, m, 32);
        const int c = __shfl(mycol, m, 32);
        const float4 hr = *(const float4*)&h[(size_t)r * 128 + lane * 4];
        const float4 hc = *(const float4*)&h[(size_t)c * 128 + lane * 4];
        *(v4s*)&feat[m * FS + lane * 4]       = pack4(hr);
        *(v4s*)&feat[m * FS + 128 + lane * 4] = pack4(hc);
    }
    __builtin_amdgcn_wave_barrier();

    // per-lane copies of radial / row-base pointers for its 8 output rows
    float  rloc[8];
    float* mrowp[8];
#pragma unroll
    for (int v = 0; v < 8; ++v) {
        rloc[v]  = radS[wid][v + 8 * half];
        mrowp[v] = mi_accum + (size_t)rowS[wid][v + 8 * half] * 128;
    }

    // ---- layer 1: edge_feat @ ew1 (+radial rank-1) + eb1, SiLU ---------
    v16bf a1[8];
#pragma unroll
    for (int kt = 0; kt < 8; ++kt) a1[kt] = ldA(feat, FS, kt, lane);

#pragma unroll
    for (int nt = 0; nt < 8; ++nt) {
        const int n = nt * 16 + mrow;
        const float b1 = eb1[n];
        const float wr = ew1r[n];
        v8f acc = {0, 0, 0, 0, 0, 0, 0, 0};
#pragma unroll
        for (int kt = 0; kt < 8; ++kt) acc = wmma_bf16(a1[kt], ldB(ew1s, kt * 8 + nt, lane), acc);
#pragma unroll
        for (int v = 0; v < 8; ++v) {
            const float t = silu(acc[v] + b1 + rloc[v] * wr);
            hid[(v + 8 * half) * HS + n] = f2bf(t);
        }
    }
    __builtin_amdgcn_wave_barrier();

    // ---- layer 2: hidden @ ew2 + eb2, SiLU -> m_ij ---------------------
    v16bf a2[4];
#pragma unroll
    for (int kt = 0; kt < 4; ++kt) a2[kt] = ldA(hid, HS, kt, lane);

#pragma unroll
    for (int nt = 0; nt < 8; ++nt) {
        const int n = nt * 16 + mrow;
        const float b2 = eb2[n];
        v8f acc = {0, 0, 0, 0, 0, 0, 0, 0};
#pragma unroll
        for (int kt = 0; kt < 4; ++kt) acc = wmma_bf16(a2[kt], ldB(ew2s, kt * 8 + nt, lane), acc);
#pragma unroll
        for (int v = 0; v < 8; ++v) {
            const float t = silu(acc[v] + b2);
            unsafeAtomicAdd(mrowp[v] + n, t);                // segment_sum(m_ij)
            hid[(v + 8 * half) * HS + n] = f2bf(t);          // reuse LDS for m_ij
        }
    }
    __builtin_amdgcn_wave_barrier();

    // ---- coord_mlp: silu(m_ij @ cw1 + cb1) . cw2 -----------------------
    v16bf a3[4];
#pragma unroll
    for (int kt = 0; kt < 4; ++kt) a3[kt] = ldA(hid, HS, kt, lane);

    float pd[8] = {0, 0, 0, 0, 0, 0, 0, 0};
#pragma unroll
    for (int nt = 0; nt < 8; ++nt) {
        const int n = nt * 16 + mrow;
        const float b3 = cb1[n];
        const float w2 = cw2[n];
        v8f acc = {0, 0, 0, 0, 0, 0, 0, 0};
#pragma unroll
        for (int kt = 0; kt < 4; ++kt) acc = wmma_bf16(a3[kt], ldB(cw1s, kt * 8 + nt, lane), acc);
#pragma unroll
        for (int v = 0; v < 8; ++v) pd[v] += silu(acc[v] + b3) * w2;
    }
    // reduce over the 16 lanes of each half-group (each group covers all n)
#pragma unroll
    for (int mask = 1; mask <= 8; mask <<= 1)
#pragma unroll
        for (int v = 0; v < 8; ++v) pd[v] += __shfl_xor(pd[v], mask, 32);

    if (mrow == 0) {
#pragma unroll
        for (int v = 0; v < 8; ++v) cwS[wid][v + 8 * half] = pd[v];
    }
    __builtin_amdgcn_wave_barrier();

    // ---- x_update scatter ---------------------------------------------
    if (lane < 16) {
        const float w = cwS[wid][lane];
        const int   r = rowS[wid][lane];
#pragma unroll
        for (int d = 0; d < 3; ++d)
            unsafeAtomicAdd(&x_accum[(size_t)r * 3 + d], diffS[wid][lane][d] * w);
    }
}

// ---------------------------------------------------------------------------
// Node kernel: h_new = h + node_mlp(concat(h, m_i)); 16 nodes/wave, 2 waves/blk
// ---------------------------------------------------------------------------
__global__ __launch_bounds__(64) void egnn_node_kernel(
    const float* __restrict__ h, const float* __restrict__ mi,
    const unsigned short* __restrict__ nw1s, const float* __restrict__ nb1,
    const unsigned short* __restrict__ nw2s, const float* __restrict__ nb2,
    float* __restrict__ out_h) {

    __shared__ unsigned short featS[2][16 * FS];
    __shared__ unsigned short hidS[2][16 * HS];

    const int lane = threadIdx.x & 31;
    const int wid  = threadIdx.x >> 5;
    const int mrow = lane & 15, half = lane >> 4;
    const int nb   = (blockIdx.x * 2 + wid) * 16;

    unsigned short* feat = featS[wid];
    unsigned short* hid  = hidS[wid];

    // stage [h | m_i] as bf16
#pragma unroll 4
    for (int m = 0; m < 16; ++m) {
        const float4 hv = *(const float4*)&h[(size_t)(nb + m) * 128 + lane * 4];
        const float4 mv = *(const float4*)&mi[(size_t)(nb + m) * 128 + lane * 4];
        *(v4s*)&feat[m * FS + lane * 4]       = pack4(hv);
        *(v4s*)&feat[m * FS + 128 + lane * 4] = pack4(mv);
    }
    __builtin_amdgcn_wave_barrier();

    v16bf a1[8];
#pragma unroll
    for (int kt = 0; kt < 8; ++kt) a1[kt] = ldA(feat, FS, kt, lane);

#pragma unroll
    for (int nt = 0; nt < 8; ++nt) {
        const int n = nt * 16 + mrow;
        const float b1 = nb1[n];
        v8f acc = {0, 0, 0, 0, 0, 0, 0, 0};
#pragma unroll
        for (int kt = 0; kt < 8; ++kt) acc = wmma_bf16(a1[kt], ldB(nw1s, kt * 8 + nt, lane), acc);
#pragma unroll
        for (int v = 0; v < 8; ++v)
            hid[(v + 8 * half) * HS + n] = f2bf(silu(acc[v] + b1));
    }
    __builtin_amdgcn_wave_barrier();

    v16bf a2[4];
#pragma unroll
    for (int kt = 0; kt < 4; ++kt) a2[kt] = ldA(hid, HS, kt, lane);

#pragma unroll
    for (int nt = 0; nt < 8; ++nt) {
        const int n = nt * 16 + mrow;
        const float b2 = nb2[n];
        v8f acc = {0, 0, 0, 0, 0, 0, 0, 0};
#pragma unroll
        for (int kt = 0; kt < 4; ++kt) acc = wmma_bf16(a2[kt], ldB(nw2s, kt * 8 + nt, lane), acc);
#pragma unroll
        for (int v = 0; v < 8; ++v) {
            const size_t node = (size_t)(nb + v + 8 * half);
            out_h[node * 128 + n] = acc[v] + b2 + h[node * 128 + n];   // residual
        }
    }
}

// ---------------------------------------------------------------------------
__global__ void egnn_x_kernel(const float* __restrict__ x, const float* __restrict__ xacc,
                              float* __restrict__ out_x, int Nn) {
    const int i = blockIdx.x * blockDim.x + threadIdx.x;
    if (i < Nn * 3) out_x[i] = x[i] + xacc[i] * (1.0f / (float)(Nn - 1));
}

// ---------------------------------------------------------------------------
static inline size_t align256(size_t v) { return (v + 255) & ~(size_t)255; }

extern "C" void kernel_launch(void* const* d_in, const int* in_sizes, int n_in,
                              void* d_out, int out_size, void* d_ws, size_t ws_size,
                              hipStream_t stream) {
    const float*     h    = (const float*)d_in[0];
    const float*     x    = (const float*)d_in[1];
    const long long* eidx = (const long long*)d_in[2];
    const float* ew1 = (const float*)d_in[3];
    const float* eb1 = (const float*)d_in[4];
    const float* ew2 = (const float*)d_in[5];
    const float* eb2 = (const float*)d_in[6];
    const float* nw1 = (const float*)d_in[7];
    const float* nb1 = (const float*)d_in[8];
    const float* nw2 = (const float*)d_in[9];
    const float* nb2 = (const float*)d_in[10];
    const float* cw1 = (const float*)d_in[11];
    const float* cb1 = (const float*)d_in[12];
    const float* cw2 = (const float*)d_in[13];

    const int Nn = in_sizes[0] / 128;      // 20000
    const int E  = in_sizes[2] / 2;        // 640000

    // workspace layout
    char*  ws   = (char*)d_ws;
    size_t off  = 0;
    float* mi_accum = (float*)(ws + off); off += align256((size_t)Nn * 128 * 4);
    float* x_accum  = (float*)(ws + off); off += align256((size_t)Nn * 3 * 4);
    const size_t accum_bytes = off;
    unsigned short* ew1s = (unsigned short*)(ws + off); off += align256((size_t)256 * 128 * 2);
    unsigned short* ew2s = (unsigned short*)(ws + off); off += align256((size_t)128 * 128 * 2);
    unsigned short* cw1s = (unsigned short*)(ws + off); off += align256((size_t)128 * 128 * 2);
    unsigned short* nw1s = (unsigned short*)(ws + off); off += align256((size_t)256 * 128 * 2);
    unsigned short* nw2s = (unsigned short*)(ws + off); off += align256((size_t)128 * 128 * 2);

    // zero segment-sum accumulators (every call; graph-capture safe)
    hipMemsetAsync(d_ws, 0, accum_bytes, stream);

    // swizzle weights -> bf16 B-operand tiles
    const int thr = 256;
    egnn_swz_kernel<<<(256 * 128 + thr - 1) / thr, thr, 0, stream>>>(ew1, ew1s, 256, 128);
    egnn_swz_kernel<<<(128 * 128 + thr - 1) / thr, thr, 0, stream>>>(ew2, ew2s, 128, 128);
    egnn_swz_kernel<<<(128 * 128 + thr - 1) / thr, thr, 0, stream>>>(cw1, cw1s, 128, 128);
    egnn_swz_kernel<<<(256 * 128 + thr - 1) / thr, thr, 0, stream>>>(nw1, nw1s, 256, 128);
    egnn_swz_kernel<<<(128 * 128 + thr - 1) / thr, thr, 0, stream>>>(nw2, nw2s, 128, 128);

    // edge pass: 16 edges/wave, 4 waves/block (E = 640000 is a multiple of 64)
    const float* ew1r = ew1 + (size_t)256 * 128;   // radial row of ew1
    egnn_edge_kernel<<<E / (16 * EW), EW * 32, 0, stream>>>(
        h, x, eidx, E, ew1s, eb1, ew2s, eb2, ew1r, cw1s, cb1, cw2, mi_accum, x_accum);

    // node pass: 16 nodes/wave, 2 waves/block (N = 20000 is a multiple of 32)
    egnn_node_kernel<<<Nn / 32, 64, 0, stream>>>(h, mi_accum, nw1s, nb1, nw2s, nb2,
                                                 (float*)d_out);

    // x finalize
    egnn_x_kernel<<<(Nn * 3 + 255) / 256, 256, 0, stream>>>(
        x, x_accum, (float*)d_out + (size_t)Nn * 128, Nn);
}
